// similarity_measure3_24240795418931
// MI455X (gfx1250) — matrix operand order repaired and checked by reading, not compiled, not run on hardware
//
#include <hip/hip_runtime.h>
#include <hip/hip_fp16.h>

typedef __attribute__((ext_vector_type(16))) _Float16 v16h;
typedef __attribute__((ext_vector_type(8)))  float    v8f;

#define HW     131072           // 256*512
#define C_IN   69
#define NPIX   524288           // 4*256*512
#define NTILES 32768            // NPIX / 16
#define NEGSLOPE 0.01f

__device__ __forceinline__ float lrelu(float v) { return v > 0.f ? v : NEGSLOPE * v; }

__global__ void sim3_init_max(unsigned* gmax) {
    if (threadIdx.x == 0 && blockIdx.x == 0) *gmax = 0u;
}

// Main fused MLP: per wave, 16-pixel tiles. Layers 1-2 via WMMA f16, tail via VALU.
__global__ __launch_bounds__(256) void sim3_mlp_kernel(
    const float* __restrict__ x,
    const float* __restrict__ w1, const float* __restrict__ w2,
    const float* __restrict__ w3, const float* __restrict__ w4,
    const float* __restrict__ w5,
    float* __restrict__ out, unsigned* __restrict__ gmax)
{
    __shared__ _Float16 lds1[8][32 * 16]; // per-wave H1, [k][n] layout (B-fragment friendly)
    __shared__ _Float16 lds2[8][16 * 16]; // per-wave H2, [n][m] layout (per-pixel tail friendly)

    const int wid  = threadIdx.x >> 5;
    const int lane = threadIdx.x & 31;
    const int n    = lane & 15;   // pixel-in-tile (B/D column)
    const int half = lane >> 4;   // lane group
    const int wgid = blockIdx.x * 8 + wid;
    const int wstride = gridDim.x * 8;

    // ---- A fragments (weights), loop-invariant.
    // A 16x32 f16, contiguous k order used here (same in B): lane m = lane&15; element e -> k = 16*half + e
    v16h a1lo[3], a1hi[3], a2;
    #pragma unroll
    for (int j = 0; j < 3; ++j) {
        #pragma unroll
        for (int e = 0; e < 16; ++e) {
            const int eo   = e;                           // 0..15 (compile-time)
            const int kmin = 32 * j + eo;                 // k when half==0 (compile-time)
            int k = kmin + 16 * half;                     // per-lane k
            float lo, hi;
            if (kmin + 16 < C_IN) {                       // valid for both halves
                lo = w1[n * C_IN + k];
                hi = w1[(16 + n) * C_IN + k];
            } else if (kmin < C_IN) {                     // valid only for half==0
                float tlo = w1[n * C_IN + kmin];          // always in-bounds address
                float thi = w1[(16 + n) * C_IN + kmin];
                lo = (half == 0) ? tlo : 0.f;             // v_cndmask, no branch
                hi = (half == 0) ? thi : 0.f;
            } else {                                      // uniformly padded
                lo = 0.f; hi = 0.f;
            }
            a1lo[j][e] = (_Float16)lo;
            a1hi[j][e] = (_Float16)hi;
        }
    }
    #pragma unroll
    for (int e = 0; e < 16; ++e) {
        int k = 16 * half + e;
        a2[e] = (_Float16)w2[n * 32 + k];
    }

    _Float16* L1 = lds1[wid];
    _Float16* L2 = lds2[wid];
    float wmax = 0.f;

    for (int tile = wgid; tile < NTILES; tile += wstride) {
        const int p0  = tile << 4;
        const int b   = p0 >> 17;              // HW = 2^17
        const int rem = (p0 & (HW - 1)) + n;
        const float* xb = x + (size_t)(b * C_IN) * (size_t)HW + rem;

        __builtin_prefetch(xb + (wstride << 4), 0, 0); // next tile, ch0 -> global_prefetch_b8

        // ---- Layer 1: D(32x16) = W1(32x69pad96) * X(96x16), 6 WMMAs
        v8f c1lo = {}; v8f c1hi = {};
        #pragma unroll
        for (int j = 0; j < 3; ++j) {
            // B 32x16 f16, contiguous k order used here (same as A): lane column n; element e -> k = 16*half + e
            v16h bfrag;
            #pragma unroll
            for (int e = 0; e < 16; ++e) {
                const int cmin = 32 * j + e;              // channel when half==0 (compile-time)
                float v;
                if (cmin + 16 < C_IN) {
                    // valid for both halves: unconditional load at per-lane address
                    v = xb[(size_t)(cmin + 16 * half) * HW];
                } else if (cmin < C_IN) {
                    // valid only for half==0: branch-free load at clamped (in-bounds)
                    // address + select-to-zero for half==1 lanes
                    float t = xb[(size_t)cmin * HW];
                    v = (half == 0) ? t : 0.f;
                } else {
                    v = 0.f;                              // uniformly padded, no load
                }
                bfrag[e] = (_Float16)v;
            }
            c1lo = __builtin_amdgcn_wmma_f32_16x16x32_f16(false, a1lo[j], false, bfrag,
                                                          (short)0, c1lo, false, false);
            c1hi = __builtin_amdgcn_wmma_f32_16x16x32_f16(false, a1hi[j], false, bfrag,
                                                          (short)0, c1hi, false, false);
        }

        // ---- lrelu + relayout D(32x16) -> LDS [k][n] for layer-2 B fragment
        // D layout: VGPR r holds row m = r + 8*half, column n.
        #pragma unroll
        for (int r = 0; r < 8; ++r) {
            int klo = r + 8 * half;
            int khi = 16 + r + 8 * half;
            L1[klo * 16 + n] = (_Float16)lrelu(c1lo[r]);
            L1[khi * 16 + n] = (_Float16)lrelu(c1hi[r]);
        }
        asm volatile("s_wait_dscnt 0x0" ::: "memory");

        // ---- Layer 2: D(16x16) = W2(16x32) * H1(32x16), 1 WMMA
        v16h b2;
        #pragma unroll
        for (int e = 0; e < 16; ++e)
            b2[e] = L1[(16 * half + e) * 16 + n];
        v8f c2 = {};
        c2 = __builtin_amdgcn_wmma_f32_16x16x32_f16(false, a2, false, b2,
                                                    (short)0, c2, false, false);

        // ---- lrelu + relayout D(16x16) -> LDS [n][m] for per-pixel tail
        #pragma unroll
        for (int r = 0; r < 8; ++r) {
            int m = r + 8 * half;
            L2[n * 16 + m] = (_Float16)lrelu(c2[r]);
        }
        asm volatile("s_wait_dscnt 0x0" ::: "memory");

        // ---- Tail layers 16->8->4->1 in VALU (164 MACs/pixel), lanes 0-15 only
        float av = 0.f;
        if (half == 0) {
            float h2[16];
            #pragma unroll
            for (int k = 0; k < 16; ++k) h2[k] = (float)L2[n * 16 + k];
            float h3[8];
            #pragma unroll
            for (int o = 0; o < 8; ++o) {
                float acc = 0.f;
                #pragma unroll
                for (int k = 0; k < 16; ++k) acc += w3[o * 16 + k] * h2[k];
                h3[o] = lrelu(acc);
            }
            float h4[4];
            #pragma unroll
            for (int o = 0; o < 4; ++o) {
                float acc = 0.f;
                #pragma unroll
                for (int k = 0; k < 8; ++k) acc += w4[o * 8 + k] * h3[k];
                h4[o] = lrelu(acc);
            }
            float v = 0.f;
            #pragma unroll
            for (int k = 0; k < 4; ++k) v += w5[k] * h4[k];
            av = fabsf(v);
            out[p0 + n] = av;          // pre-normalization |v|
        }

        // wave max (lanes 16-31 contribute 0.0, safe since av >= 0)
        #pragma unroll
        for (int off = 16; off > 0; off >>= 1)
            av = fmaxf(av, __shfl_xor(av, off, 32));
        wmax = fmaxf(wmax, av);
    }

    // |v| >= 0 so float ordering == unsigned bit ordering
    if (lane == 0) atomicMax(gmax, __float_as_uint(wmax));
}

__global__ void sim3_finalize(float* __restrict__ out, const unsigned* __restrict__ gmax, int nelem) {
    int i = blockIdx.x * blockDim.x + threadIdx.x;
    if (i < nelem) {
        float mx = __uint_as_float(*gmax);
        out[i] = expf(-out[i] / mx);
    }
}

extern "C" void kernel_launch(void* const* d_in, const int* in_sizes, int n_in,
                              void* d_out, int out_size, void* d_ws, size_t ws_size,
                              hipStream_t stream) {
    (void)in_sizes; (void)n_in; (void)out_size; (void)ws_size;
    const float* x  = (const float*)d_in[0];
    const float* w1 = (const float*)d_in[1];
    const float* w2 = (const float*)d_in[2];
    const float* w3 = (const float*)d_in[3];
    const float* w4 = (const float*)d_in[4];
    const float* w5 = (const float*)d_in[5];
    float* out = (float*)d_out;
    unsigned* gmax = (unsigned*)d_ws;

    sim3_init_max<<<1, 64, 0, stream>>>(gmax);
    // 256 blocks * 8 waves = 2048 waves; 32768 tiles -> exactly 16 tiles/wave
    sim3_mlp_kernel<<<256, 256, 0, stream>>>(x, w1, w2, w3, w4, w5, out, gmax);
    sim3_finalize<<<(NPIX + 255) / 256, 256, 0, stream>>>(out, gmax, NPIX);
}
